// Attention4DDownsample_53523882443539
// MI455X (gfx1250) — matrix operand order repaired and checked by reading, not compiled, not run on hardware
//
#include <hip/hip_runtime.h>
#include <hip/hip_bf16.h>

typedef __attribute__((ext_vector_type(16))) _Float16 v16h;
typedef __attribute__((ext_vector_type(8)))  float    v8f;

#define BB 256
#define CC 384
#define NN 196
#define N2 49
#define HEADS 8
#define KD 16
#define DD 64
#define NHKD 128
#define DH 512
#define OUTC 384

// ---------------- WMMA helpers (CDNA5 layouts, wave32) ----------------

__device__ __forceinline__ v8f wmma_f16(v16h a, v16h b, v8f c) {
  // D = A(16x32 f16) * B(32x16 f16) + C(16x16 f32)
  return __builtin_amdgcn_wmma_f32_16x16x32_f16(
      /*neg_a=*/false, a, /*neg_b=*/false, b,
      /*c_mod=*/(short)0, c, /*reuse_a=*/false, /*reuse_b=*/false);
}

// A fragment: 16x32 f16, row-major in LDS, As[m*lda + k].
// lanes 0-15: M=lane, VGPR0..7 -> K 0..15
// lanes 16-31: M=lane-16, VGPR0..7 -> K 16..31
// Each lane = one contiguous 32B run.
__device__ __forceinline__ v16h load_frag_A(const _Float16* As, int lda) {
  int lane  = threadIdx.x & 31;
  int m     = lane & 15;
  int khalf = lane >> 4;
  v16h a;
#pragma unroll
  for (int i = 0; i < 8; ++i) {
    int k = khalf * 16 + i * 2;
    a[2 * i]     = As[m * lda + k];
    a[2 * i + 1] = As[m * lda + k + 1];
  }
  return a;
}

// B fragment from N-major ("transposed") tile Bt[n][k], ldk halves per row.
// lane n holds K kbase..kbase+15 for its column: one contiguous 32B run.
__device__ __forceinline__ v16h load_frag_Bt(const _Float16* Bt, int ldk) {
  int lane = threadIdx.x & 31;
  const _Float16* p = Bt + (lane & 15) * ldk + (lane >> 4) * 16;
  return *(const v16h*)p;  // 2x ds_load_b128 (32B-aligned in all uses)
}

// ---------------- Kernel 1: fold BN into f16 weights (chunk-transposed) ----------------
// Wkv_t: [12][640][32]  (kc, o, k)   element = W[kc*32+k][o]
// Wq_t : [12][128][32]
// Wp_t : [16][384][32]

__global__ __launch_bounds__(256) void prep_kernel(
    const float* kw, const float* kb, const float* kbs, const float* kbb,
    const float* vw, const float* vb, const float* vbs, const float* vbb,
    const float* qw, const float* qb, const float* qbs, const float* qbb,
    const float* pw, const float* pb, const float* pbs, const float* pbb,
    _Float16* Wkv, float* bkv, _Float16* Wq, float* bq,
    _Float16* Wp, float* bp) {
  int idx = blockIdx.x * 256 + threadIdx.x;
  const int SZ_KV = 12 * 640 * 32;   // 245760
  const int SZ_Q  = 12 * 128 * 32;   // 49152
  const int SZ_P  = 16 * 384 * 32;   // 196608
  if (idx < SZ_KV) {
    int kc = idx / (640 * 32);
    int r  = idx - kc * (640 * 32);
    int o = r >> 5, k = r & 31;
    int c = kc * 32 + k;
    float v = (o < NHKD) ? kw[o * CC + c] * kbs[o]
                         : vw[(o - NHKD) * CC + c] * vbs[o - NHKD];
    Wkv[idx] = (_Float16)v;
    return;
  }
  idx -= SZ_KV;
  if (idx < 640) {
    bkv[idx] = (idx < NHKD) ? kb[idx] * kbs[idx] + kbb[idx]
                            : vb[idx - NHKD] * vbs[idx - NHKD] + vbb[idx - NHKD];
    return;
  }
  idx -= 640;
  if (idx < SZ_Q) {
    int kc = idx / (128 * 32);
    int r  = idx - kc * (128 * 32);
    int o = r >> 5, k = r & 31;
    int c = kc * 32 + k;
    Wq[idx] = (_Float16)(qw[o * CC + c] * qbs[o]);
    return;
  }
  idx -= SZ_Q;
  if (idx < NHKD) { bq[idx] = qb[idx] * qbs[idx] + qbb[idx]; return; }
  idx -= NHKD;
  if (idx < SZ_P) {
    int kc = idx / (384 * 32);
    int r  = idx - kc * (384 * 32);
    int o = r >> 5, k = r & 31;
    int c = kc * 32 + k;
    Wp[idx] = (_Float16)(pw[o * DH + c] * pbs[o]);
    return;
  }
  idx -= SZ_P;
  if (idx < OUTC) { bp[idx] = pb[idx] * pbs[idx] + pbb[idx]; }
}

// ---------------- Kernel 2: q input = dw3x3s2(x)+bias + pool(x), f16 ----------------

__global__ __launch_bounds__(256) void qin_kernel(
    const float* x, const float* qlw, const float* qlb, _Float16* qin) {
  int idx = blockIdx.x * 256 + threadIdx.x;  // [b][c][p], p in 49
  int b = idx / (CC * N2);
  int rem = idx - b * (CC * N2);
  int c = rem / N2, p = rem - c * N2;
  int oh = p / 7, ow = p - oh * 7;
  const float* xp = x + (b * CC + c) * NN;
  float s = qlb[c];
#pragma unroll
  for (int kh = 0; kh < 3; ++kh) {
#pragma unroll
    for (int kw2 = 0; kw2 < 3; ++kw2) {
      int ih = oh * 2 - 1 + kh, iw = ow * 2 - 1 + kw2;
      if (ih >= 0 && ih < 14 && iw >= 0 && iw < 14)
        s += xp[ih * 14 + iw] * qlw[c * 9 + kh * 3 + kw2];
    }
  }
  s += xp[(oh * 2) * 14 + (ow * 2)];  // pool_q
  qin[idx] = (_Float16)s;
}

// ---------------- Kernel 3: k|v GEMM, M=16 spatial, N=640, K=384 ----------------

__global__ __launch_bounds__(256) void kv_gemm_kernel(
    const float* x, const _Float16* Wkv, const float* bkv,
    _Float16* kh, _Float16* vh, float* v4) {
  __shared__ _Float16 xs[16 * 32];                   // A tile [m][k]
  __shared__ alignas(16) _Float16 ws[640 * 32];      // B chunk [o][k] (N-major)
  const int t = threadIdx.x, wave = t >> 5, lane = t & 31;
  const int m0 = blockIdx.x * 16;

  v8f zero = {};
  v8f acc[5];
#pragma unroll
  for (int j = 0; j < 5; ++j) acc[j] = zero;

  for (int kc = 0; kc < 12; ++kc) {
    // A staging: consecutive threads -> consecutive spatial (coalesced)
    for (int e = t; e < 512; e += 256) {
      int m = e & 15, k = e >> 4;
      int g = m0 + m, b = g / NN, n = g - b * NN;
      xs[m * 32 + k] = (_Float16)x[(b * CC + kc * 32 + k) * NN + n];
    }
    const uint4* src = (const uint4*)(Wkv + kc * 640 * 32);
    uint4* dst = (uint4*)ws;
    for (int e = t; e < 2560; e += 256) {
      __builtin_prefetch(src + e + 2560, 0, 1);  // next chunk -> global_prefetch
      dst[e] = src[e];
    }
    __syncthreads();
    // Batch all fragment loads, then issue WMMAs back-to-back
    // (independent accumulators -> no RAW hazard stalls between them).
    v16h a = load_frag_A(xs, 32);
    v16h bf[5];
#pragma unroll
    for (int j = 0; j < 5; ++j)
      bf[j] = load_frag_Bt(ws + (wave * 5 + j) * 16 * 32, 32);
#pragma unroll
    for (int j = 0; j < 5; ++j)
      acc[j] = wmma_f16(a, bf[j], acc[j]);
    __syncthreads();
  }

  int rl0 = (lane >> 4) * 8;
#pragma unroll
  for (int j = 0; j < 5; ++j) {
    int o = (wave * 5 + j) * 16 + (lane & 15);
#pragma unroll
    for (int r = 0; r < 8; ++r) {
      int g = m0 + rl0 + r, b = g / NN, n = g - b * NN;
      float val = acc[j][r] + bkv[o];
      if (o < NHKD) {  // k: [b][h][kd][n]
        kh[((b * HEADS + (o >> 4)) * KD + (o & 15)) * NN + n] = (_Float16)val;
      } else {
        int ov = o - NHKD;  // v: [b][h][n][d] f16 + v4 f32 for depthwise
        vh[((b * HEADS + (ov >> 6)) * NN + n) * DD + (ov & 63)] = (_Float16)val;
        v4[(b * DH + ov) * NN + n] = val;
      }
    }
  }
}

// ---------------- Kernel 4: q GEMM, M=16 over B*49, N=128, K=384 ----------------

__global__ __launch_bounds__(256) void q_gemm_kernel(
    const _Float16* qin, const _Float16* Wq, const float* bq, _Float16* qh) {
  __shared__ _Float16 xs[16 * 32];
  __shared__ alignas(16) _Float16 ws[128 * 32];
  const int t = threadIdx.x, wave = t >> 5, lane = t & 31;
  const int m0 = blockIdx.x * 16;
  v8f acc = {};

  for (int kc = 0; kc < 12; ++kc) {
    for (int e = t; e < 512; e += 256) {
      int m = e & 15, k = e >> 4;
      int g = m0 + m, b = g / N2, p = g - b * N2;
      xs[m * 32 + k] = qin[(b * CC + kc * 32 + k) * N2 + p];
    }
    const uint4* src = (const uint4*)(Wq + kc * 128 * 32);
    uint4* dst = (uint4*)ws;
    for (int e = t; e < 512; e += 256) dst[e] = src[e];
    __syncthreads();
    v16h a  = load_frag_A(xs, 32);
    v16h bf = load_frag_Bt(ws + wave * 16 * 32, 32);
    acc = wmma_f16(a, bf, acc);
    __syncthreads();
  }

  int o = wave * 16 + (lane & 15);
  int h = o >> 4, kd = o & 15;
  int rl0 = (lane >> 4) * 8;
#pragma unroll
  for (int r = 0; r < 8; ++r) {
    int g = m0 + rl0 + r, b = g / N2, p = g - b * N2;
    qh[((b * HEADS + h) * N2 + p) * KD + kd] = (_Float16)(acc[r] + bq[o]);
  }
}

// ---------------- Kernel 5: v_local depthwise 3x3 s2 + BN ----------------

__global__ __launch_bounds__(256) void vlocal_kernel(
    const float* v4, const float* vlw, const float* vlb,
    const float* vls, const float* vlbb, float* vloc) {
  int idx = blockIdx.x * 256 + threadIdx.x;  // [b][ch][p]
  int b = idx / (DH * N2);
  int rem = idx - b * (DH * N2);
  int ch = rem / N2, p = rem - ch * N2;
  int oh = p / 7, ow = p - oh * 7;
  const float* vp = v4 + (b * DH + ch) * NN;
  float s = vlb[ch];
#pragma unroll
  for (int kh = 0; kh < 3; ++kh) {
#pragma unroll
    for (int kw2 = 0; kw2 < 3; ++kw2) {
      int ih = oh * 2 - 1 + kh, iw = ow * 2 - 1 + kw2;
      if (ih >= 0 && ih < 14 && iw >= 0 && iw < 14)
        s += vp[ih * 14 + iw] * vlw[ch * 9 + kh * 3 + kw2];
    }
  }
  vloc[idx] = s * vls[ch] + vlbb[ch];
}

// ---------------- Kernel 6: attention per (b,h): QK + softmax + PV ----------------

__global__ __launch_bounds__(256) void attn_kernel(
    const _Float16* qh, const _Float16* kh, const _Float16* vh,
    const float* bias_tab, float* ao) {
  __shared__ _Float16 qs[64 * 32];             // q padded: 64 rows x K32
  __shared__ alignas(32) _Float16 ks[208 * 32];// k N-major: [n][k32]
  __shared__ alignas(16) char scps[64 * 832];  // scores f32 [64][208] / probs f16 (416-stride)
  __shared__ alignas(32) _Float16 vs[64 * 224];// v N-major: [d][k224]
  float*    scf = (float*)scps;                // lda 208 floats (832B rows)
  _Float16* psh = (_Float16*)scps;             // lda 416 halves (same 832B rows)

  const int t = threadIdx.x, wave = t >> 5, lane = t & 31;
  const int b = blockIdx.x >> 3, h = blockIdx.x & 7;
  const _Float16* qbase = qh + ((b * HEADS + h) * N2) * KD;
  const _Float16* kbase = kh + ((b * HEADS + h) * KD) * NN;
  const _Float16* vbase = vh + ((b * HEADS + h) * NN) * DD;
  const float*    btab  = bias_tab + h * N2 * NN;

  for (int e = t; e < 64 * 32; e += 256) {
    int m = e >> 5, k = e & 31;
    qs[e] = (m < N2 && k < KD) ? qbase[m * KD + k] : (_Float16)0.f;
  }
  for (int e = t; e < 208 * 32; e += 256) {   // ks[n][k]
    int n = e >> 5, k = e & 31;
    ks[e] = (k < KD && n < NN) ? kbase[k * NN + n] : (_Float16)0.f;
  }
  __syncthreads();

  // Phase 1: scores = q*k^T * 0.25 + bias (K padded 16->32)
  {
    int mt = wave >> 1;
    v16h a = load_frag_A(qs + mt * 16 * 32, 32);
    int rl0 = (lane >> 4) * 8;
    for (int nt = (wave & 1); nt < 13; nt += 2) {
      v16h bf = load_frag_Bt(ks + nt * 16 * 32, 32);
      v8f c = {};
      c = wmma_f16(a, bf, c);
      int col = nt * 16 + (lane & 15);
#pragma unroll
      for (int r = 0; r < 8; ++r) {
        int row = mt * 16 + rl0 + r;
        float bias = (row < N2 && col < NN) ? btab[row * NN + col] : 0.f;
        scf[row * 208 + col] = c[r] * 0.25f + bias;
      }
    }
  }
  __syncthreads();

  // Phase 2: softmax over 196 keys (wave per row; in-place f32 -> f16, row-private)
  for (int row = wave; row < N2; row += 8) {
    float vreg[7];
    float mx = -3.0e38f;
#pragma unroll
    for (int j = 0; j < 7; ++j) {
      int col = lane + 32 * j;
      float v = (col < NN) ? scf[row * 208 + col] : -3.0e38f;
      vreg[j] = v;
      mx = fmaxf(mx, v);
    }
    for (int off = 16; off > 0; off >>= 1) mx = fmaxf(mx, __shfl_xor(mx, off, 32));
    float s = 0.f;
#pragma unroll
    for (int j = 0; j < 7; ++j) {
      int col = lane + 32 * j;
      float e = (col < NN) ? __expf(vreg[j] - mx) : 0.f;
      vreg[j] = e;
      s += e;
    }
    for (int off = 16; off > 0; off >>= 1) s += __shfl_xor(s, off, 32);
    float inv = 1.f / s;
    _Float16* prow = psh + row * 416;
#pragma unroll
    for (int j = 0; j < 7; ++j) prow[lane + 32 * j] = (_Float16)(vreg[j] * inv);
  }
  __syncthreads();

  for (int e = t; e < 64 * 224; e += 256) {   // vs[d][k]
    int d = e / 224, k = e - d * 224;
    vs[e] = (k < NN) ? vbase[k * DD + d] : (_Float16)0.f;
  }
  __syncthreads();

  // Phase 3: out = P(64x224) * V(224x64), 7 K-chunks of 32.
  // A (probs) fragment shared by the wave's two output tiles per chunk.
  {
    int mt  = wave >> 1;
    int nt0 = (wave & 1) * 2, nt1 = nt0 + 1;
    int rl0 = (lane >> 4) * 8;
    v8f c0 = {}, c1 = {};
    for (int kc = 0; kc < 7; ++kc) {
      v16h a  = load_frag_A(psh + mt * 16 * 416 + kc * 32, 416);
      v16h b0 = load_frag_Bt(vs + nt0 * 16 * 224 + kc * 32, 224);
      v16h b1 = load_frag_Bt(vs + nt1 * 16 * 224 + kc * 32, 224);
      c0 = wmma_f16(a, b0, c0);
      c1 = wmma_f16(a, b1, c1);
    }
    int d0 = nt0 * 16 + (lane & 15);
    int d1 = nt1 * 16 + (lane & 15);
#pragma unroll
    for (int r = 0; r < 8; ++r) {
      int p = mt * 16 + rl0 + r;
      if (p < N2) {
        ao[(b * DH + h * DD + d0) * N2 + p] = c0[r];
        ao[(b * DH + h * DD + d1) * N2 + p] = c1[r];
      }
    }
  }
}

// ---------------- Kernel 7: out = Wp * relu(ao + vloc) + bias ----------------

__global__ __launch_bounds__(256) void pout_gemm_kernel(
    const float* ao, const float* vloc, const _Float16* Wp, const float* bp,
    float* out) {
  __shared__ _Float16 xs[16 * 32];
  __shared__ alignas(16) _Float16 ws[384 * 32];
  const int t = threadIdx.x, wave = t >> 5, lane = t & 31;
  const int m0 = blockIdx.x * 16;
  v8f zero = {};
  v8f acc[3];
#pragma unroll
  for (int j = 0; j < 3; ++j) acc[j] = zero;

  for (int kc = 0; kc < 16; ++kc) {
    for (int e = t; e < 512; e += 256) {
      int m = e & 15, k = e >> 4;
      int g = m0 + m, b = g / N2, p = g - b * N2;
      int ch = kc * 32 + k;
      float v = ao[(b * DH + ch) * N2 + p] + vloc[(b * DH + ch) * N2 + p];
      xs[m * 32 + k] = (_Float16)fmaxf(v, 0.f);
    }
    const uint4* src = (const uint4*)(Wp + kc * 384 * 32);
    uint4* dst = (uint4*)ws;
    for (int e = t; e < 1536; e += 256) dst[e] = src[e];
    __syncthreads();
    v16h a = load_frag_A(xs, 32);
    v16h bf[3];
#pragma unroll
    for (int j = 0; j < 3; ++j)
      bf[j] = load_frag_Bt(ws + (wave * 3 + j) * 16 * 32, 32);
#pragma unroll
    for (int j = 0; j < 3; ++j)
      acc[j] = wmma_f16(a, bf[j], acc[j]);
    __syncthreads();
  }

  int rl0 = (lane >> 4) * 8;
#pragma unroll
  for (int j = 0; j < 3; ++j) {
    int o = (wave * 3 + j) * 16 + (lane & 15);
#pragma unroll
    for (int r = 0; r < 8; ++r) {
      int g = m0 + rl0 + r, b = g / N2, p = g - b * N2;
      out[(b * OUTC + o) * N2 + p] = acc[j][r] + bp[o];
    }
  }
}

// ---------------- Launch ----------------

static inline size_t align256(size_t x) { return (x + 255) & ~size_t(255); }

extern "C" void kernel_launch(void* const* d_in, const int* in_sizes, int n_in,
                              void* d_out, int out_size, void* d_ws, size_t ws_size,
                              hipStream_t stream) {
  const float* x    = (const float*)d_in[0];
  const float* qlw  = (const float*)d_in[1];
  const float* qlb  = (const float*)d_in[2];
  const float* qpw  = (const float*)d_in[3];
  const float* qpb  = (const float*)d_in[4];
  const float* qbs  = (const float*)d_in[5];
  const float* qbb  = (const float*)d_in[6];
  const float* kw   = (const float*)d_in[7];
  const float* kb   = (const float*)d_in[8];
  const float* kbs  = (const float*)d_in[9];
  const float* kbb  = (const float*)d_in[10];
  const float* vw   = (const float*)d_in[11];
  const float* vb   = (const float*)d_in[12];
  const float* vbs  = (const float*)d_in[13];
  const float* vbb  = (const float*)d_in[14];
  const float* vlw  = (const float*)d_in[15];
  const float* vlb  = (const float*)d_in[16];
  const float* vls  = (const float*)d_in[17];
  const float* vlbb = (const float*)d_in[18];
  const float* btab = (const float*)d_in[19];
  const float* pw   = (const float*)d_in[20];
  const float* pb   = (const float*)d_in[21];
  const float* pbs  = (const float*)d_in[22];
  const float* pbb  = (const float*)d_in[23];
  float* out = (float*)d_out;

  char* p = (char*)d_ws;
  auto carve = [&](size_t bytes) { char* r = p; p += align256(bytes); return r; };
  _Float16* Wkv  = (_Float16*)carve(sizeof(_Float16) * CC * 640);
  float*    bkv  = (float*)   carve(sizeof(float) * 640);
  _Float16* Wq   = (_Float16*)carve(sizeof(_Float16) * CC * NHKD);
  float*    bq   = (float*)   carve(sizeof(float) * NHKD);
  _Float16* Wp   = (_Float16*)carve(sizeof(_Float16) * DH * OUTC);
  float*    bp   = (float*)   carve(sizeof(float) * OUTC);
  _Float16* qin  = (_Float16*)carve(sizeof(_Float16) * (size_t)BB * CC * N2);
  _Float16* qh   = (_Float16*)carve(sizeof(_Float16) * (size_t)BB * NHKD * N2);
  _Float16* kh   = (_Float16*)carve(sizeof(_Float16) * (size_t)BB * NHKD * NN);
  _Float16* vh   = (_Float16*)carve(sizeof(_Float16) * (size_t)BB * DH * NN);
  float*    v4   = (float*)   carve(sizeof(float) * (size_t)BB * DH * NN);
  float*    vloc = (float*)   carve(sizeof(float) * (size_t)BB * DH * N2);
  float*    ao   = (float*)   carve(sizeof(float) * (size_t)BB * DH * N2);

  prep_kernel<<<1925, 256, 0, stream>>>(kw, kb, kbs, kbb, vw, vb, vbs, vbb,
                                        qpw, qpb, qbs, qbb, pw, pb, pbs, pbb,
                                        Wkv, bkv, Wq, bq, Wp, bp);
  qin_kernel<<<(BB * CC * N2) / 256, 256, 0, stream>>>(x, qlw, qlb, qin);
  kv_gemm_kernel<<<(BB * NN) / 16, 256, 0, stream>>>(x, Wkv, bkv, kh, vh, v4);
  q_gemm_kernel<<<(BB * N2) / 16, 256, 0, stream>>>(qin, Wq, bq, qh);
  vlocal_kernel<<<(BB * DH * N2) / 256, 256, 0, stream>>>(v4, vlw, vlb, vls, vlbb, vloc);
  attn_kernel<<<BB * HEADS, 256, 0, stream>>>(qh, kh, vh, btab, ao);
  pout_gemm_kernel<<<(BB * N2) / 16, 256, 0, stream>>>(ao, vloc, Wp, bp, out);
}
